// BatchedDCRNN_28458453303763
// MI455X (gfx1250) — matrix operand order repaired and checked by reading, not compiled, not run on hardware
//
#include <hip/hip_runtime.h>
#include <hip/hip_bf16.h>
#include <cstddef>
#include <cstdint>

// ---------------- problem constants (match reference) ----------------
constexpr int kB   = 64;     // batch
constexpr int kT   = 12;     // timesteps
constexpr int kN   = 325;    // nodes
constexpr int kFin = 32;     // input feats
constexpr int kFo  = 128;    // output feats
constexpr int kC   = 160;    // kFin + kFo (gate input width)
constexpr int kM   = kB * kN;        // 20800 GEMM rows
constexpr int kKC  = 5 * kC;         // 800 concatenated-K

typedef __attribute__((ext_vector_type(16))) __bf16 v16bf;
typedef __attribute__((ext_vector_type(8)))  __bf16 v8bf;
typedef __attribute__((ext_vector_type(8)))  float  v8f;

__device__ __forceinline__ unsigned short f2bf(float f) {
  unsigned u = __float_as_uint(f);
  u += 0x7FFFu + ((u >> 16) & 1u);          // round-to-nearest-even
  return (unsigned short)(u >> 16);
}

// ---------------- setup kernels (deterministic, no atomics) ----------------

// deg_out[d] = sum of w over edges with row==d ; deg_in[d] = sum over col==d
__global__ void dcrnn_deg(const int* __restrict__ row, const int* __restrict__ col,
                          const float* __restrict__ w, int E,
                          float* __restrict__ degO, float* __restrict__ degI) {
  int d = blockIdx.x * blockDim.x + threadIdx.x;
  if (d >= kN) return;
  float so = 0.f, si = 0.f;
  for (int e = 0; e < E; ++e) {
    if (row[e] == d) so += w[e];
    if (col[e] == d) si += w[e];
  }
  degO[d] = so; degI[d] = si;
}

// stable argsort of key = col*N + row : perm[rank(e)] = e
__global__ void dcrnn_perm(const int* __restrict__ row, const int* __restrict__ col,
                           int E, int* __restrict__ perm) {
  int e = blockIdx.x * blockDim.x + threadIdx.x;
  if (e >= E) return;
  int key = col[e] * kN + row[e];
  int rank = 0;
  for (int e2 = 0; e2 < E; ++e2) {
    int k2 = col[e2] * kN + row[e2];
    rank += (k2 < key) || (k2 == key && e2 < e);
  }
  perm[rank] = e;
}

// fw[e] = 1/deg_out[row[e]]  (norm_out)
__global__ void dcrnn_fw(const int* __restrict__ row, const float* __restrict__ degO,
                         int E, float* __restrict__ fw) {
  int e = blockIdx.x * blockDim.x + threadIdx.x;
  if (e < E) fw[e] = 1.f / degO[row[e]];
}

// rev edge list: rsrc[j]=col[perm[j]], rdst[j]=row[perm[j]], rw[j]=1/deg_in[row[j]]
// (norm_in indexed in ORIGINAL order j — faithful to the reference quirk)
__global__ void dcrnn_rev(const int* __restrict__ row, const int* __restrict__ col,
                          const int* __restrict__ perm, const float* __restrict__ degI,
                          int E, int* __restrict__ rsrc, int* __restrict__ rdst,
                          float* __restrict__ rw) {
  int j = blockIdx.x * blockDim.x + threadIdx.x;
  if (j >= E) return;
  int pidx = perm[j];
  rsrc[j] = col[pidx];
  rdst[j] = row[pidx];
  rw[j]   = 1.f / degI[row[j]];
}

// ofs[d] = #edges with dst < d   (d in [0, N]; ofs[N] = E)
__global__ void dcrnn_ofs(const int* __restrict__ dst, int E, int* __restrict__ ofs) {
  int d = blockIdx.x * blockDim.x + threadIdx.x;
  if (d > kN) return;
  int c = 0;
  for (int e = 0; e < E; ++e) c += (dst[e] < d);
  ofs[d] = c;
}

// stable CSR fill keyed by dst
__global__ void dcrnn_csr(const int* __restrict__ src, const int* __restrict__ dst,
                          const float* __restrict__ w, int E,
                          int* __restrict__ srcs, float* __restrict__ ws) {
  int e = blockIdx.x * blockDim.x + threadIdx.x;
  if (e >= E) return;
  int d = dst[e], pos = 0;
  for (int e2 = 0; e2 < E; ++e2) {
    int d2 = dst[e2];
    pos += (d2 < d) + ((d2 == d) && (e2 < e));
  }
  srcs[pos] = src[e];
  ws[pos]   = w[e];
}

// W [2][3][160][128] fp32 -> WT [128][800] bf16 (K concat order:
// t0 = W[0][0]+W[1][0], t1 = W[0][1], t2 = W[1][1], t3 = W[0][2], t4 = W[1][2])
__global__ void dcrnn_wconv(const float* __restrict__ W, unsigned short* __restrict__ WT) {
  int idx = blockIdx.x * blockDim.x + threadIdx.x;
  if (idx >= kFo * kKC) return;
  int n = idx / kKC, kc = idx % kKC;
  int t = kc / kC, c = kc % kC;
  auto at = [&](int i, int k) { return W[(((size_t)i * 3 + k) * kC + c) * kFo + n]; };
  float v;
  if      (t == 0) v = at(0, 0) + at(1, 0);
  else if (t == 1) v = at(0, 1);
  else if (t == 2) v = at(1, 1);
  else if (t == 3) v = at(0, 2);
  else             v = at(1, 2);
  WT[idx] = f2bf(v);
}

__global__ void dcrnn_zero(float* __restrict__ ptr, int n) {
  int i = blockIdx.x * blockDim.x + threadIdx.x;
  if (i < n) ptr[i] = 0.f;
}

// ---------------- per-timestep kernels ----------------

// XH[row][0:32] = X[b,t,n,:] ; XH[row][32:160] = H (optionally * R)
__global__ void dcrnn_concat(const float* __restrict__ X, int t,
                             const float* __restrict__ H, const float* __restrict__ R,
                             float* __restrict__ XH, int useR) {
  int idx = blockIdx.x * blockDim.x + threadIdx.x;
  if (idx >= kM * kC) return;
  int rowi = idx / kC, c = idx % kC;
  int b = rowi / kN, n = rowi % kN;
  float v;
  if (c < kFin) {
    v = X[(((size_t)b * kT + t) * kN + n) * kFin + c];
  } else {
    size_t ho = (size_t)rowi * kFo + (c - kFin);
    v = H[ho];
    if (useR) v *= R[ho];
  }
  XH[idx] = v;
}

// out[b,d,c] = (useSub ? 2*g - sub[b,d,c] : g),  g = sum_j ws[j]*Xin[b,srcs[j],c]
__global__ void dcrnn_gather(const float* __restrict__ Xin, const int* __restrict__ ofs,
                             const int* __restrict__ srcs, const float* __restrict__ ws,
                             const float* __restrict__ sub, float* __restrict__ outp,
                             int useSub) {
  int d = blockIdx.x;        // node
  int b = blockIdx.y;        // batch
  int c = threadIdx.x;       // 0..159
  int s = ofs[d], e = ofs[d + 1];
  float acc = 0.f;
  for (int j = s; j < e; ++j)
    acc += ws[j] * Xin[((size_t)b * kN + srcs[j]) * kC + c];
  size_t o = ((size_t)b * kN + d) * kC + c;
  outp[o] = useSub ? (2.f * acc - sub[o]) : acc;
}

// ---------------- WMMA gate GEMM: out = act( [A0|A1|A2|A3|A4] @ WT^T + bias ) ---
// A*: [kM][160] fp32 ; WT: [128][800] bf16 (N-major, K contiguous)
// block = 256 thr (8 waves); block tile 128x128; wave tile 16x128 (8 n-tiles).
// Double-buffered LDS; B tiles staged with GLOBAL_LOAD_ASYNC_TO_LDS_B128 (ASYNCcnt),
// A tiles staged via float4 loads + fp32->bf16 convert + ds_store.
__global__ __launch_bounds__(256, 2)
void dcrnn_gate_gemm(const float* __restrict__ A0, const float* __restrict__ A1,
                     const float* __restrict__ A2, const float* __restrict__ A3,
                     const float* __restrict__ A4,
                     const unsigned short* __restrict__ WT,
                     const float* __restrict__ bias,
                     float* __restrict__ outp, int act) {
  __shared__ unsigned short As[2][128][40];   // 2 x (128 rows x 32 k bf16), skewed
  __shared__ unsigned short Bs[2][128][40];   // 2 x (128 n    x 32 k bf16)
  const int tid  = threadIdx.x;
  const int wave = tid >> 5, lane = tid & 31;
  const int r15  = lane & 15, half = lane >> 4;
  const int kb   = half * 8;               // A-fragment K base per half-wave
  const int row0 = blockIdx.x * 128;

  v8f acc[8];
#pragma unroll
  for (int i = 0; i < 8; ++i) acc[i] = (v8f){0.f, 0.f, 0.f, 0.f, 0.f, 0.f, 0.f, 0.f};

  auto stage = [&](int ks, int buf) {
    const int sel = ks / 5;
    const int k0  = (ks - sel * 5) * 32;
    const float* __restrict__ A =
        sel == 0 ? A0 : sel == 1 ? A1 : sel == 2 ? A2 : sel == 3 ? A3 : A4;
    // A block: 128 x 32 fp32 -> bf16 LDS
#pragma unroll
    for (int it = 0; it < 4; ++it) {
      int i  = tid + it * 256;
      int rr = i >> 3;
      int cc = (i & 7) << 2;
      int grow = row0 + rr;
      float4 v = make_float4(0.f, 0.f, 0.f, 0.f);
      if (grow < kM) v = *(const float4*)(A + (size_t)grow * kC + k0 + cc);
      As[buf][rr][cc + 0] = f2bf(v.x);
      As[buf][rr][cc + 1] = f2bf(v.y);
      As[buf][rr][cc + 2] = f2bf(v.z);
      As[buf][rr][cc + 3] = f2bf(v.w);
    }
    // B block: 128 x 32 bf16, DMA'd straight into LDS (async, no VGPR roundtrip)
#pragma unroll
    for (int it = 0; it < 2; ++it) {
      int i  = tid + it * 256;
      int nn = i >> 2;
      int cc = (i & 3) << 3;
      unsigned ldsoff = (unsigned)(uintptr_t)&Bs[buf][nn][cc];
      const unsigned short* gsrc = WT + (size_t)nn * kKC + ks * 32 + cc;
      asm volatile("global_load_async_to_lds_b128 %0, %1, off"
                   :: "v"(ldsoff), "v"(gsrc) : "memory");
    }
  };

  // prologue
  stage(0, 0);
  asm volatile("s_wait_asynccnt 0x0" ::: "memory");
  __syncthreads();

  for (int ks = 0; ks < 25; ++ks) {
    const int cur = ks & 1;
    if (ks + 1 < 25) stage(ks + 1, cur ^ 1);   // overlap staging with WMMA

    // A fragment (ISA 16-bit A 16x32 layout: half-wave K 0-7 & 16-23 / 8-15 & 24-31)
    const int arow = wave * 16 + r15;
    v8bf alo = *(const v8bf*)&As[cur][arow][kb];
    v8bf ahi = *(const v8bf*)&As[cur][arow][kb + 16];
    v16bf af = __builtin_shufflevector(alo, ahi, 0, 1, 2, 3, 4, 5, 6, 7,
                                                 8, 9, 10, 11, 12, 13, 14, 15);
#pragma unroll
    for (int nt = 0; nt < 8; ++nt) {
      const int bn = nt * 16 + r15;
      // B fragment (32x16: half-wave K 0-7 & 16-23 / 8-15 & 24-31, matching A)
      v8bf blo = *(const v8bf*)&Bs[cur][bn][half * 8];
      v8bf bhi = *(const v8bf*)&Bs[cur][bn][half * 8 + 16];
      v16bf bf = __builtin_shufflevector(blo, bhi, 0, 1, 2, 3, 4, 5, 6, 7,
                                                   8, 9, 10, 11, 12, 13, 14, 15);
      acc[nt] = __builtin_amdgcn_wmma_f32_16x16x32_bf16(
          false, af, false, bf, (short)0, acc[nt], false, false);
    }
    // async B staging must land before anyone reads the next buffer
    asm volatile("s_wait_asynccnt 0x0" ::: "memory");
    __syncthreads();
  }

  // epilogue: bias + activation (0=sigmoid, 1=tanh); C/D layout: VGPR j -> M=j(+8)
#pragma unroll
  for (int nt = 0; nt < 8; ++nt) {
    const int coln = nt * 16 + r15;
    const float bv = bias[coln];
#pragma unroll
    for (int j = 0; j < 8; ++j) {
      const int grow = row0 + wave * 16 + half * 8 + j;
      if (grow < kM) {
        float v = acc[nt][j] + bv;
        v = act ? tanhf(v) : 1.f / (1.f + __expf(-v));
        outp[(size_t)grow * kFo + coln] = v;
      }
    }
  }
}

// GRU update: Hn = Z*H + (1-Z)*Ht ; H <- Hn ; out[b,t,n,:] = Hn
__global__ void dcrnn_update(const float* __restrict__ Z, const float* __restrict__ Ht,
                             float* __restrict__ H, float* __restrict__ outp, int t) {
  int idx = blockIdx.x * blockDim.x + threadIdx.x;
  if (idx >= kM * kFo) return;
  float z = Z[idx], h = H[idx], ht = Ht[idx];
  float hn = z * h + (1.f - z) * ht;
  H[idx] = hn;
  int rowi = idx / kFo, c = idx % kFo;
  int b = rowi / kN, n = rowi % kN;
  outp[(((size_t)b * kT + t) * kN + n) * kFo + c] = hn;
}

// ---------------- driver ----------------
extern "C" void kernel_launch(void* const* d_in, const int* in_sizes, int n_in,
                              void* d_out, int out_size, void* d_ws, size_t ws_size,
                              hipStream_t stream) {
  const float* X  = (const float*)d_in[0];
  const int*   EI = (const int*)d_in[1];
  const float* EW = (const float*)d_in[2];
  const float* Wz = (const float*)d_in[3];
  const float* bz = (const float*)d_in[4];
  const float* Wr = (const float*)d_in[5];
  const float* br = (const float*)d_in[6];
  const float* Wh = (const float*)d_in[7];
  const float* bh = (const float*)d_in[8];
  const int E = in_sizes[2];
  const int* row = EI;
  const int* col = EI + E;
  float* out = (float*)d_out;

  char* p = (char*)d_ws;
  auto bump = [&](size_t bytes) -> char* {
    char* r = p;
    p += (bytes + 255) & ~(size_t)255;
    return r;
  };
  float* XH   = (float*)bump((size_t)kM * kC * 4);
  float* TxO  = (float*)bump((size_t)kM * kC * 4);
  float* TxI  = (float*)bump((size_t)kM * kC * 4);
  float* TxO2 = (float*)bump((size_t)kM * kC * 4);
  float* TxI2 = (float*)bump((size_t)kM * kC * 4);
  float* Zb   = (float*)bump((size_t)kM * kFo * 4);
  float* Rb   = (float*)bump((size_t)kM * kFo * 4);
  float* Htb  = (float*)bump((size_t)kM * kFo * 4);
  float* Hb   = (float*)bump((size_t)kM * kFo * 4);
  float* degO = (float*)bump(kN * 4);
  float* degI = (float*)bump(kN * 4);
  float* fw   = (float*)bump((size_t)E * 4);
  int*   perm = (int*)bump((size_t)E * 4);
  int*   rsrc = (int*)bump((size_t)E * 4);
  int*   rdst = (int*)bump((size_t)E * 4);
  float* rw   = (float*)bump((size_t)E * 4);
  int*   ofsF = (int*)bump((kN + 1) * 4);
  int*   ofsR = (int*)bump((kN + 1) * 4);
  int*   srcF = (int*)bump((size_t)E * 4);
  float* wF   = (float*)bump((size_t)E * 4);
  int*   srcR = (int*)bump((size_t)E * 4);
  float* wR   = (float*)bump((size_t)E * 4);
  unsigned short* WTz = (unsigned short*)bump((size_t)kFo * kKC * 2);
  unsigned short* WTr = (unsigned short*)bump((size_t)kFo * kKC * 2);
  unsigned short* WTh = (unsigned short*)bump((size_t)kFo * kKC * 2);

  const int eBlk = (E + 255) / 256;

  // ---- setup ----
  dcrnn_deg<<<(kN + 255) / 256, 256, 0, stream>>>(row, col, EW, E, degO, degI);
  dcrnn_perm<<<eBlk, 256, 0, stream>>>(row, col, E, perm);
  dcrnn_fw<<<eBlk, 256, 0, stream>>>(row, degO, E, fw);
  dcrnn_rev<<<eBlk, 256, 0, stream>>>(row, col, perm, degI, E, rsrc, rdst, rw);
  dcrnn_ofs<<<(kN + 256) / 256, 256, 0, stream>>>(col, E, ofsF);
  dcrnn_ofs<<<(kN + 256) / 256, 256, 0, stream>>>(rdst, E, ofsR);
  dcrnn_csr<<<eBlk, 256, 0, stream>>>(row, col, fw, E, srcF, wF);
  dcrnn_csr<<<eBlk, 256, 0, stream>>>(rsrc, rdst, rw, E, srcR, wR);
  dcrnn_wconv<<<(kFo * kKC + 255) / 256, 256, 0, stream>>>(Wz, WTz);
  dcrnn_wconv<<<(kFo * kKC + 255) / 256, 256, 0, stream>>>(Wr, WTr);
  dcrnn_wconv<<<(kFo * kKC + 255) / 256, 256, 0, stream>>>(Wh, WTh);
  dcrnn_zero<<<(kM * kFo + 255) / 256, 256, 0, stream>>>(Hb, kM * kFo);

  const dim3 gGather(kN, kB);
  const int gemmBlocks = (kM + 127) / 128;
  const int catBlocks  = (kM * kC + 255) / 256;
  const int updBlocks  = (kM * kFo + 255) / 256;

  // ---- recurrence ----
  for (int t = 0; t < kT; ++t) {
    // XH = [x_t, H]
    dcrnn_concat<<<catBlocks, 256, 0, stream>>>(X, t, Hb, nullptr, XH, 0);
    // diffusion hops
    dcrnn_gather<<<gGather, kC, 0, stream>>>(XH, ofsF, srcF, wF, nullptr, TxO, 0);
    dcrnn_gather<<<gGather, kC, 0, stream>>>(XH, ofsR, srcR, wR, nullptr, TxI, 0);
    dcrnn_gather<<<gGather, kC, 0, stream>>>(TxO, ofsF, srcF, wF, XH, TxO2, 1);
    dcrnn_gather<<<gGather, kC, 0, stream>>>(TxI, ofsR, srcR, wR, XH, TxI2, 1);
    // Z, R gates (sigmoid)
    dcrnn_gate_gemm<<<gemmBlocks, 256, 0, stream>>>(XH, TxO, TxI, TxO2, TxI2,
                                                    WTz, bz, Zb, 0);
    dcrnn_gate_gemm<<<gemmBlocks, 256, 0, stream>>>(XH, TxO, TxI, TxO2, TxI2,
                                                    WTr, br, Rb, 0);
    // XH2 = [x_t, H*R] and its hops
    dcrnn_concat<<<catBlocks, 256, 0, stream>>>(X, t, Hb, Rb, XH, 1);
    dcrnn_gather<<<gGather, kC, 0, stream>>>(XH, ofsF, srcF, wF, nullptr, TxO, 0);
    dcrnn_gather<<<gGather, kC, 0, stream>>>(XH, ofsR, srcR, wR, nullptr, TxI, 0);
    dcrnn_gather<<<gGather, kC, 0, stream>>>(TxO, ofsF, srcF, wF, XH, TxO2, 1);
    dcrnn_gather<<<gGather, kC, 0, stream>>>(TxI, ofsR, srcR, wR, XH, TxI2, 1);
    // candidate gate (tanh)
    dcrnn_gate_gemm<<<gemmBlocks, 256, 0, stream>>>(XH, TxO, TxI, TxO2, TxI2,
                                                    WTh, bh, Htb, 1);
    // GRU update + emit
    dcrnn_update<<<updBlocks, 256, 0, stream>>>(Zb, Htb, Hb, out, t);
  }
}
